// Attention_28003186770488
// MI455X (gfx1250) — matrix operand order repaired and checked by reading, not compiled, not run on hardware
//
#include <hip/hip_runtime.h>

// ============================================================================
// CDNA5 (gfx1250) fused attention block, f16 WMMA pipeline, fp32 accumulate.
// Round 4: GEMM LDS staging via GLOBAL_LOAD_ASYNC_TO_LDS_B128 (ASYNCcnt path);
// builtin signature per clang diagnostic: (AS1 v4i*, AS3 v4i*, Iint, Iint).
// ============================================================================

typedef __attribute__((ext_vector_type(16))) _Float16 v16h;
typedef __attribute__((ext_vector_type(8)))  float    v8f;
typedef __attribute__((ext_vector_type(4)))  int      v4i;

#define WMMA(a, b, c) __builtin_amdgcn_wmma_f32_16x16x32_f16( \
    false, (a), false, (b), (short)0, (c), false, false)

#if __has_builtin(__builtin_amdgcn_global_load_async_to_lds_b128)
#define USE_ASYNC_LDS 1
#define AS1 __attribute__((address_space(1)))
#define AS3 __attribute__((address_space(3)))
__device__ __forceinline__ void async_b128(const _Float16* gsrc, _Float16* ldst) {
    __builtin_amdgcn_global_load_async_to_lds_b128(
        (AS1 v4i*)(v4i*)gsrc, (AS3 v4i*)(v4i*)ldst, 0, 0);
}
#else
#define USE_ASYNC_LDS 0
#endif

__device__ __forceinline__ void wait_async0() {
#if __has_builtin(__builtin_amdgcn_s_wait_asynccnt)
    __builtin_amdgcn_s_wait_asynccnt(0);
#else
    asm volatile("s_wait_asynccnt 0x0" ::: "memory");
#endif
}

// ---------------------------------------------------------------------------
// Fragment loaders per ISA 7.12.2 (wave32), fully vectorized (2x 16B each).
// A (16x32 f16): lane m = lane&15; lane group selects K-halves {0|8}:
//   halves [0..7] = row[khi .. khi+7], halves [8..15] = row[16+khi .. 16+khi+7]
__device__ __forceinline__ v16h frag_a(const _Float16* base, int strideM) {
    int lane = threadIdx.x & 31;
    const _Float16* row = base + (size_t)(lane & 15) * strideM + ((lane >> 4) << 3);
    union { uint4 u[2]; v16h h; } t;
    t.u[0] = *(const uint4*)(row);
    t.u[1] = *(const uint4*)(row + 16);
    return t.h;
}

// B (32x16 f16): lane n = lane&15; lane group selects K-halves {0|8}, same as A:
// halves [0..7] = K[khi..khi+7], [8..15] = K[16+khi..16+khi+7]. Source must be K-contiguous per column:
// elem(k,n) at base[n*strideN + k].
__device__ __forceinline__ v16h frag_bk(const _Float16* base, int strideN) {
    int lane = threadIdx.x & 31;
    const _Float16* p = base + (size_t)(lane & 15) * strideN + ((lane >> 4) << 3);
    union { uint4 u[2]; v16h h; } t;
    t.u[0] = *(const uint4*)(p);
    t.u[1] = *(const uint4*)(p + 16);
    return t.h;
}
// C/D (16x16 f32): VGPR r, lane: row = 8*(lane>>4) + r, col = lane&15.

// ---------------------------------------------------------------------------
// LDS-staged GEMM wave tile: C(128x64) block, BK=32, 256 thr (8 waves),
// wave (wm,wn) in 4x2 grid computes 32x32 via 2x2 C-fragments.
// A: row-major (M x K).  Bt: TRANSPOSED (N x K) so fragments are K-contiguous.
#define BM 128
#define BN 64
#define BK 32
#define AS_LD 40   // padded stride (halves): 80B rows -> 16B aligned, bank-clean
#define BS_LD 40

__device__ __forceinline__ void gemm_wave_tile(
    const _Float16* __restrict__ A, const _Float16* __restrict__ Bt,
    int lda, int ldbt, int m0, int n0, int K,
    _Float16* As, _Float16* Bs, v8f acc[2][2]) {
    int tid = threadIdx.x;
    int w = tid >> 5, wm = w >> 1, wn = w & 1;
    for (int k0 = 0; k0 < K; k0 += BK) {
#if USE_ASYNC_LDS
        // async global -> LDS staging (ASYNCcnt), no VGPR round-trip
#pragma unroll
        for (int u = 0; u < 2; ++u) {
            int li  = tid * 2 + u;
            int row = li >> 2;
            int col = (li & 3) << 3;
            async_b128(A + (size_t)(m0 + row) * lda + k0 + col, As + row * AS_LD + col);
        }
        {
            int row = tid >> 2;
            int col = (tid & 3) << 3;
            async_b128(Bt + (size_t)(n0 + row) * ldbt + k0 + col, Bs + row * BS_LD + col);
        }
        wait_async0();
#else
        // stage A tile 128x32: 512 x uint4, 2 per thread
#pragma unroll
        for (int u = 0; u < 2; ++u) {
            int li  = tid * 2 + u;
            int row = li >> 2;
            int col = (li & 3) << 3;
            *(uint4*)(As + row * AS_LD + col) =
                *(const uint4*)(A + (size_t)(m0 + row) * lda + k0 + col);
        }
        // stage Bt tile 64x32 (rows = output cols, K contiguous): 256 x uint4
        {
            int row = tid >> 2;
            int col = (tid & 3) << 3;
            *(uint4*)(Bs + row * BS_LD + col) =
                *(const uint4*)(Bt + (size_t)(n0 + row) * ldbt + k0 + col);
        }
#endif
        if (k0 + BK < K) {  // speculative prefetch of next Bt tile
            __builtin_prefetch(Bt + (size_t)(n0 + (tid >> 2)) * ldbt + k0 + BK + ((tid & 3) << 3), 0, 1);
        }
        __syncthreads();
        v16h av[2], bv[2];
        av[0] = frag_a(As + (size_t)(wm * 32) * AS_LD, AS_LD);
        av[1] = frag_a(As + (size_t)(wm * 32 + 16) * AS_LD, AS_LD);
        bv[0] = frag_bk(Bs + (size_t)(wn * 32) * BS_LD, BS_LD);
        bv[1] = frag_bk(Bs + (size_t)(wn * 32 + 16) * BS_LD, BS_LD);
#pragma unroll
        for (int sm = 0; sm < 2; ++sm)
#pragma unroll
            for (int sn = 0; sn < 2; ++sn)
                acc[sm][sn] = WMMA(av[sm], bv[sn], acc[sm][sn]);
        __syncthreads();
    }
}

// ---------------------------------------------------------------------------
// Conversion kernels (weights written TRANSPOSED: out[n][k] = W[k][n])
__global__ __launch_bounds__(256) void f32_to_f16_kernel(const float* __restrict__ s,
                                                         _Float16* __restrict__ d, int n) {
    int i = blockIdx.x * 256 + threadIdx.x;
    if (i < n) d[i] = (_Float16)s[i];
}

__global__ __launch_bounds__(256) void pack_wqkv_t_kernel(const float* __restrict__ Wq,
                                                          const float* __restrict__ Wkv,
                                                          _Float16* __restrict__ wt, int n) {
    int i = blockIdx.x * 256 + threadIdx.x;
    if (i < n) {
        int ncol = i >> 10, k = i & 1023;   // wt is (3072 x 1024)
        float v = (ncol < 1024) ? Wq[(size_t)k * 1024 + ncol]
                                : Wkv[(size_t)k * 2048 + (ncol - 1024)];
        wt[i] = (_Float16)v;
    }
}

__global__ __launch_bounds__(256) void cvt_wo_t_kernel(const float* __restrict__ Wo,
                                                       _Float16* __restrict__ wt, int n) {
    int i = blockIdx.x * 256 + threadIdx.x;
    if (i < n) {
        int ncol = i >> 10, k = i & 1023;   // wt is (1024 x 1024) transposed
        wt[i] = (_Float16)Wo[(size_t)k * 1024 + ncol];
    }
}

// ---------------------------------------------------------------------------
// Kernel 1: QKV projection (4096x1024 @ 1024x3072) + RoPE epilogue.
// q gets 1/sqrt(64) folded in. q,k -> (b,h,n,d) f16;  v -> (b,h,d,n) f16 (V^T).
__global__ __launch_bounds__(256) void qkv_rope_kernel(
    const _Float16* __restrict__ xh, const _Float16* __restrict__ wqkvt,
    _Float16* __restrict__ qh, _Float16* __restrict__ kh, _Float16* __restrict__ vt) {
    __shared__ __align__(16) _Float16 As[BM * AS_LD];
    __shared__ __align__(16) _Float16 Bs[BN * BS_LD];
    int m0 = blockIdx.x * BM, n0 = blockIdx.y * BN;
    v8f zero = {0.f, 0.f, 0.f, 0.f, 0.f, 0.f, 0.f, 0.f};
    v8f acc[2][2] = {{zero, zero}, {zero, zero}};
    gemm_wave_tile(xh, wqkvt, 1024, 1024, m0, n0, 1024, As, Bs, acc);

    int w = threadIdx.x >> 5, wm = w >> 1, wn = w & 1;
    int lane = threadIdx.x & 31, nloc = lane & 15, mh = (lane >> 4) << 3;
#pragma unroll
    for (int sm = 0; sm < 2; ++sm)
#pragma unroll
        for (int sn = 0; sn < 2; ++sn) {
            int col     = n0 + wn * 32 + sn * 16 + nloc;
            int rowbase = m0 + wm * 32 + sm * 16 + mh;
            if (col < 2048) {  // q or k -> RoPE
                int dd = col & 63;
                int fi = dd >> 1;
                // inv_freq = theta^(-2*fi/64);  ln(10000) = 9.2103403...
                float inv = __expf(-(float)(2 * fi) * (9.210340371976184f / 64.f));
                _Float16* dst   = (col < 1024) ? qh : kh;
                float postscale = (col < 1024) ? 0.125f : 1.0f;  // fold 1/sqrt(d) into q
                int hh = (col & 1023) >> 6;
#pragma unroll
                for (int r = 0; r < 8; ++r) {
                    int row = rowbase + r;
                    int bb = row >> 11, nn = row & 2047;
                    float v  = acc[sm][sn][r];
                    float vp = __shfl_xor(v, 1, 32);  // rotate-half partner (col^1)
                    float ang = (float)nn * inv;
                    float cv = __cosf(ang), sv = __sinf(ang);
                    float o = (dd & 1) ? (v * cv + vp * sv) : (v * cv - vp * sv);
                    dst[(((size_t)(bb * 16 + hh)) * 2048 + nn) * 64 + dd] =
                        (_Float16)(o * postscale);
                }
            } else {  // v: store transposed (b,h,d,n)
                int cc = col - 2048, hh = cc >> 6, dd = cc & 63;
#pragma unroll
                for (int r = 0; r < 8; ++r) {
                    int row = rowbase + r;
                    int bb = row >> 11, nn = row & 2047;
                    vt[(((size_t)(bb * 16 + hh)) * 64 + dd) * 2048 + nn] =
                        (_Float16)acc[sm][sn][r];
                }
            }
        }
}

// ---------------------------------------------------------------------------
// Kernel 2: flash attention. grid = (32 bh, 32 q-blocks of 64), 128 thr (4 waves),
// each wave owns 16 q rows; kv tiles of 32; online softmax; P transposed
// C-layout -> A-layout through a private 16x32 f16 LDS buffer.
// K stored (n,d): Q*K^T B-frags are d-contiguous.  V stored (d,n): kv-contiguous.
__global__ __launch_bounds__(128) void attn_kernel(
    const _Float16* __restrict__ qh, const _Float16* __restrict__ kh,
    const _Float16* __restrict__ vt, _Float16* __restrict__ aoh) {
    __shared__ __align__(16) _Float16 pbuf[4 * 16 * 32];
    int bh = blockIdx.x;
    int w = threadIdx.x >> 5, lane = threadIdx.x & 31;
    int nloc = lane & 15, mh = (lane >> 4) << 3;
    const _Float16* qp = qh + (size_t)bh * 2048 * 64;
    const _Float16* kp = kh + (size_t)bh * 2048 * 64;
    const _Float16* vp = vt + (size_t)bh * 2048 * 64;
    int q0 = blockIdx.y * 64 + w * 16;

    v16h qf0 = frag_a(qp + (size_t)q0 * 64, 64);        // d = 0..31
    v16h qf1 = frag_a(qp + (size_t)q0 * 64 + 32, 64);   // d = 32..63
    v8f zero = {0.f, 0.f, 0.f, 0.f, 0.f, 0.f, 0.f, 0.f};
    v8f o[4] = {zero, zero, zero, zero};
    float mrow[8], lrow[8];
#pragma unroll
    for (int r = 0; r < 8; ++r) { mrow[r] = -1e30f; lrow[r] = 0.f; }
    _Float16* pl = pbuf + w * (16 * 32);

    for (int kv0 = 0; kv0 < 2048; kv0 += 32) {
        // S = Q * K^T : B elem(k=d, n=kv) = K[(kv0+n)*64 + k]  (d contiguous)
        v8f s0 = zero, s1 = zero;
        s0 = WMMA(qf0, frag_bk(kp + (size_t)kv0 * 64, 64), s0);
        s0 = WMMA(qf1, frag_bk(kp + (size_t)kv0 * 64 + 32, 64), s0);
        s1 = WMMA(qf0, frag_bk(kp + (size_t)(kv0 + 16) * 64, 64), s1);
        s1 = WMMA(qf1, frag_bk(kp + (size_t)(kv0 + 16) * 64 + 32, 64), s1);

        // online softmax: per-row max/sum; rows live in 16-lane groups
        float alpha[8];
#pragma unroll
        for (int r = 0; r < 8; ++r) {
            float t = fmaxf(s0[r], s1[r]);
#pragma unroll
            for (int msk = 1; msk < 16; msk <<= 1)
                t = fmaxf(t, __shfl_xor(t, msk, 32));
            float nm = fmaxf(mrow[r], t);
            alpha[r] = __expf(mrow[r] - nm);
            mrow[r]  = nm;
        }
#pragma unroll
        for (int r = 0; r < 8; ++r) {
            float p0 = __expf(s0[r] - mrow[r]);
            float p1 = __expf(s1[r] - mrow[r]);
            pl[(mh + r) * 32 + nloc]      = (_Float16)p0;   // C-layout -> row-major LDS
            pl[(mh + r) * 32 + 16 + nloc] = (_Float16)p1;
            float ps = p0 + p1;
#pragma unroll
            for (int msk = 1; msk < 16; msk <<= 1)
                ps += __shfl_xor(ps, msk, 32);
            lrow[r] = lrow[r] * alpha[r] + ps;
        }
#pragma unroll
        for (int dblk = 0; dblk < 4; ++dblk)
#pragma unroll
            for (int r = 0; r < 8; ++r) o[dblk][r] *= alpha[r];

        // O += P * V : B elem(k=kv, n=d) = V^T[(dblk*16+n)*2048 + kv0 + k]
        v16h pa = frag_a(pl, 32);
#pragma unroll
        for (int dblk = 0; dblk < 4; ++dblk)
            o[dblk] = WMMA(pa, frag_bk(vp + (size_t)(dblk * 16) * 2048 + kv0, 2048), o[dblk]);
    }

    // normalize + store to (b, n, h*d) f16
    int bb = bh >> 4, hh = bh & 15;
#pragma unroll
    for (int dblk = 0; dblk < 4; ++dblk)
#pragma unroll
        for (int r = 0; r < 8; ++r) {
            int qpos = q0 + mh + r;
            aoh[((size_t)(bb * 2048 + qpos)) * 1024 + hh * 64 + dblk * 16 + nloc] =
                (_Float16)(o[dblk][r] / lrow[r]);
        }
}

// ---------------------------------------------------------------------------
// Kernel 3: output projection (4096x1024 @ 1024x1024, Wo transposed) + bias.
__global__ __launch_bounds__(256) void out_proj_kernel(
    const _Float16* __restrict__ aoh, const _Float16* __restrict__ wot,
    const float* __restrict__ bo, float* __restrict__ out) {
    __shared__ __align__(16) _Float16 As[BM * AS_LD];
    __shared__ __align__(16) _Float16 Bs[BN * BS_LD];
    int m0 = blockIdx.x * BM, n0 = blockIdx.y * BN;
    v8f zero = {0.f, 0.f, 0.f, 0.f, 0.f, 0.f, 0.f, 0.f};
    v8f acc[2][2] = {{zero, zero}, {zero, zero}};
    gemm_wave_tile(aoh, wot, 1024, 1024, m0, n0, 1024, As, Bs, acc);

    int w = threadIdx.x >> 5, wm = w >> 1, wn = w & 1;
    int lane = threadIdx.x & 31, nloc = lane & 15, mh = (lane >> 4) << 3;
#pragma unroll
    for (int sm = 0; sm < 2; ++sm)
#pragma unroll
        for (int sn = 0; sn < 2; ++sn) {
            int col     = n0 + wn * 32 + sn * 16 + nloc;
            int rowbase = m0 + wm * 32 + sm * 16 + mh;
            float bias = bo[col];
#pragma unroll
            for (int r = 0; r < 8; ++r)
                out[(size_t)(rowbase + r) * 1024 + col] = acc[sm][sn][r] + bias;
        }
}

// ---------------------------------------------------------------------------
extern "C" void kernel_launch(void* const* d_in, const int* in_sizes, int n_in,
                              void* d_out, int out_size, void* d_ws, size_t ws_size,
                              hipStream_t stream) {
    (void)in_sizes; (void)n_in; (void)out_size; (void)ws_size;
    const float* x   = (const float*)d_in[0];   // (2,2048,1024)
    const float* Wq  = (const float*)d_in[1];   // (1024,1024)
    const float* Wkv = (const float*)d_in[2];   // (1024,2048)
    const float* Wo  = (const float*)d_in[3];   // (1024,1024)
    const float* bo  = (const float*)d_in[4];   // (1024,)
    float* out = (float*)d_out;

    // workspace layout (f16), 40 MB total; attention output reuses the x buffer
    char* ws = (char*)d_ws;
    _Float16* xh     = (_Float16*)(ws);                        // 4096x1024    (8 MB)
    _Float16* wqkvt  = (_Float16*)(ws + ((size_t)8  << 20));   // 3072x1024 ^T (6 MB)
    _Float16* wot    = (_Float16*)(ws + ((size_t)14 << 20));   // 1024x1024 ^T (2 MB)
    _Float16* qh     = (_Float16*)(ws + ((size_t)16 << 20));   // (b,h,n,d)    (8 MB)
    _Float16* kh     = (_Float16*)(ws + ((size_t)24 << 20));   // (b,h,n,d)    (8 MB)
    _Float16* vt     = (_Float16*)(ws + ((size_t)32 << 20));   // (b,h,d,n)    (8 MB)
    _Float16* aoh    = xh;                                     // reuse after QKV GEMM

    // 0) conversions (weights transposed so WMMA B-fragments are K-contiguous)
    f32_to_f16_kernel<<<(4096 * 1024 + 255) / 256, 256, 0, stream>>>(x, xh, 4096 * 1024);
    pack_wqkv_t_kernel<<<(1024 * 3072 + 255) / 256, 256, 0, stream>>>(Wq, Wkv, wqkvt, 1024 * 3072);
    cvt_wo_t_kernel<<<(1024 * 1024 + 255) / 256, 256, 0, stream>>>(Wo, wot, 1024 * 1024);

    // 1) QKV projection + RoPE   (M=4096, N=3072)
    qkv_rope_kernel<<<dim3(4096 / BM, 3072 / BN), 256, 0, stream>>>(xh, wqkvt, qh, kh, vt);

    // 2) flash attention          (32 bh-pairs x 32 q-blocks)
    attn_kernel<<<dim3(32, 2048 / 64), 128, 0, stream>>>(qh, kh, vt, aoh);

    // 3) output projection + bias (M=4096, N=1024)
    out_proj_kernel<<<dim3(4096 / BM, 1024 / BN), 256, 0, stream>>>(aoh, wot, bo, out);
}
